// MultiHeadSelfAttention_28544352650046
// MI455X (gfx1250) — compile-verified, matrix operand order checked by reading, not run on hardware
//
#include <hip/hip_runtime.h>

// ---------------------------------------------------------------------------
// Types
// ---------------------------------------------------------------------------
typedef __attribute__((ext_vector_type(16))) __bf16        v16bf;
typedef __attribute__((ext_vector_type(8)))  float         v8f;
typedef __attribute__((ext_vector_type(4)))  unsigned int  u32x4;
typedef __attribute__((ext_vector_type(8)))  unsigned int  u32x8;
typedef __attribute__((ext_vector_type(2)))  unsigned int  u32x2;
typedef __attribute__((ext_vector_type(4)))  float         f32x4;

// round-to-nearest-even f32 -> bf16 (bit twiddle; no reliance on __bf16 arith)
static __device__ __forceinline__ __bf16 tobf(float x) {
  unsigned u = __builtin_bit_cast(unsigned, x);
  u = (u + 0x7FFFu + ((u >> 16) & 1u)) >> 16;
  unsigned short s = (unsigned short)u;
  return __builtin_bit_cast(__bf16, s);
}

// Load one 16x32 bf16 A/B WMMA fragment from a row-major bf16 matrix.
// Per CDNA5 ISA 16-bit A layout: lane m (m<16) holds row m, K {0..7, 16..23};
// lane m+16 holds row m, K {8..15, 24..31}. B fragments use the identical
// per-lane pattern with "row" = C-column (NT GEMM symmetry).
static __device__ __forceinline__ v16bf load_frag(const __bf16* p, int stride, int lane) {
  const int m = lane & 15, h = lane >> 4;
  const __bf16* q = p + (size_t)m * (size_t)stride + h * 8;
  u32x4 lo = *(const u32x4*)(q);        // K chunk {0..7}/{8..15}
  u32x4 hi = *(const u32x4*)(q + 16);   // K chunk {16..23}/{24..31}
  u32x8 w = __builtin_shufflevector(lo, hi, 0, 1, 2, 3, 4, 5, 6, 7);
  return __builtin_bit_cast(v16bf, w);
}

static __device__ __forceinline__ v8f wmma_bf16(v16bf a, v16bf b, v8f c) {
  return __builtin_amdgcn_wmma_f32_16x16x32_bf16(false, a, false, b, (short)0, c,
                                                 false, false);
}

// CDNA5 async global->LDS copy (ASYNCcnt-tracked, no VGPR round trip).
// 16 bytes per lane; LDS dest address from VGPR, 64-bit global addr from VGPR pair.
static __device__ __forceinline__ void async_g2l_b128(void* lds_dst, const void* gsrc) {
  unsigned ldsa = (unsigned)(size_t)lds_dst;             // low 32 bits = LDS offset
  unsigned long long ga = (unsigned long long)(size_t)gsrc;
  asm volatile("global_load_async_to_lds_b128 %0, %1, off"
               :: "v"(ldsa), "v"(ga)
               : "memory");
}

// Wait until at most N async copies remain in flight (in-order completion =>
// all older copies have landed in LDS).
template <int N>
static __device__ __forceinline__ void wait_async() {
#if __has_builtin(__builtin_amdgcn_s_wait_asynccnt)
  __builtin_amdgcn_s_wait_asynccnt(N);
#else
  if constexpr (N == 0)
    asm volatile("s_wait_asynccnt 0x0" ::: "memory");
  else if constexpr (N == 2)
    asm volatile("s_wait_asynccnt 0x2" ::: "memory");
  else
    asm volatile("s_wait_asynccnt 0x6" ::: "memory");
#endif
}

// ---------------------------------------------------------------------------
// Kernel 1: f32 -> bf16 convert (4 elements / thread)
// ---------------------------------------------------------------------------
__global__ __launch_bounds__(256) void cvt_f32_bf16(const float* __restrict__ src,
                                                    __bf16* __restrict__ dst, int n4) {
  int i = blockIdx.x * 256 + threadIdx.x;
  if (i >= n4) return;
  f32x4 f = *(const f32x4*)(src + (size_t)i * 4);
  u32x2 o;
  o[0] = (unsigned)__builtin_bit_cast(unsigned short, tobf(f[0])) |
         ((unsigned)__builtin_bit_cast(unsigned short, tobf(f[1])) << 16);
  o[1] = (unsigned)__builtin_bit_cast(unsigned short, tobf(f[2])) |
         ((unsigned)__builtin_bit_cast(unsigned short, tobf(f[3])) << 16);
  *(u32x2*)(dst + (size_t)i * 4) = o;
}

// ---------------------------------------------------------------------------
// Kernels 2 & 4: NT bf16 GEMM  C(M x Ncols) = A(M x 1024) * B(Ncols x 1024)^T
// Workgroup tile 64(M) x 128(N), 8 waves (2x4), wave tile 32x32 (2x2 WMMA
// tiles). K streamed in 64-wide chunks through DOUBLE-BUFFERED LDS using
// async global->LDS copies: chunk i+1 streams in while WMMAs run on chunk i.
// MODE 0: QKV projection epilogue (bias + de-interleave + scale Q + V^T store)
// MODE 1: output projection epilogue (bias, f32 store to d_out)
// ---------------------------------------------------------------------------
static constexpr int LDST = 72;  // ushort stride = 144 bytes (conflict-free)
static constexpr int ASZ = 64 * LDST;
static constexpr int BSZ = 128 * LDST;

template <int MODE>
__global__ __launch_bounds__(256) void gemm_nt(
    const __bf16* __restrict__ A, const __bf16* __restrict__ B,
    const float* __restrict__ bias, float* __restrict__ outF,
    __bf16* __restrict__ Qb, __bf16* __restrict__ Kb, __bf16* __restrict__ Vtb) {
  __shared__ __align__(16) __bf16 sA[2 * ASZ];
  __shared__ __align__(16) __bf16 sB[2 * BSZ];

  const int tid = threadIdx.x;
  const int lane = tid & 31;
  const int w = tid >> 5;
  const int wm = w & 1, wn = w >> 1;
  const int row0 = blockIdx.y * 64;
  const int col0 = blockIdx.x * 128;

  // 6 async copies per thread per K-chunk (2 for A, 4 for B).
  auto stage = [&](int buf, int kb) {
#pragma unroll
    for (int s = 0; s < 2; ++s) {  // A tile: 64 rows x 64 K = 512 x 16B
      int idx = tid + s * 256;
      int r = idx >> 3, c = (idx & 7) * 8;
      async_g2l_b128(sA + buf * ASZ + r * LDST + c,
                     A + (size_t)(row0 + r) * 1024 + kb + c);
    }
#pragma unroll
    for (int s = 0; s < 4; ++s) {  // B tile: 128 rows x 64 K
      int idx = tid + s * 256;
      int r = idx >> 3, c = (idx & 7) * 8;
      async_g2l_b128(sB + buf * BSZ + r * LDST + c,
                     B + (size_t)(col0 + r) * 1024 + kb + c);
    }
  };

  v8f acc[2][2] = {};

  stage(0, 0);
  for (int kb = 0; kb < 1024; kb += 64) {
    const int cur = (kb >> 6) & 1;
    if (kb + 64 < 1024) {
      stage(cur ^ 1, kb + 64);  // prefetch next chunk into other buffer
      wait_async<6>();          // ensure chunk `cur` has landed
    } else {
      wait_async<0>();
    }
    __syncthreads();

    const __bf16* cA = sA + cur * ASZ;
    const __bf16* cB = sB + cur * BSZ;
#pragma unroll
    for (int kk = 0; kk < 64; kk += 32) {
      v16bf af[2], bfr[2];
#pragma unroll
      for (int i = 0; i < 2; ++i)
        af[i] = load_frag(cA + (wm * 32 + i * 16) * LDST + kk, LDST, lane);
#pragma unroll
      for (int j = 0; j < 2; ++j)
        bfr[j] = load_frag(cB + (wn * 32 + j * 16) * LDST + kk, LDST, lane);
#pragma unroll
      for (int i = 0; i < 2; ++i)
#pragma unroll
        for (int j = 0; j < 2; ++j)
          acc[i][j] = wmma_bf16(af[i], bfr[j], acc[i][j]);
    }
    __syncthreads();  // all waves done with buffer `cur` before it is re-staged
  }

  // Epilogue. C layout: VGPR v -> row (v + half*8), col = lane%16 within tile.
  const int m = lane & 15, half = lane >> 4;
#pragma unroll
  for (int i = 0; i < 2; ++i) {
#pragma unroll
    for (int j = 0; j < 2; ++j) {
      const int f = col0 + wn * 32 + j * 16 + m;
      const float bv = bias[f];
#pragma unroll
      for (int v = 0; v < 8; ++v) {
        const int r = row0 + wm * 32 + i * 16 + v + half * 8;
        const float val = acc[i][j][v] + bv;
        if (MODE == 1) {
          outF[(size_t)r * 1024 + f] = val;
        } else {
          // r = l*4 + n ; f = 3*ii + jj ; ii = hh*64 + dd ; b = n*16 + hh
          const int l = r >> 2, n = r & 3;
          const int jj = f % 3, ii = f / 3;
          const int hh = ii >> 6, dd = ii & 63;
          const int b = n * 16 + hh;
          if (jj == 0)
            Qb[((size_t)b * 2048 + l) * 64 + dd] = tobf(val * 0.125f);  // fold scale
          else if (jj == 1)
            Kb[((size_t)b * 2048 + l) * 64 + dd] = tobf(val);
          else
            Vtb[((size_t)b * 64 + dd) * 2048 + l] = tobf(val);  // V transposed
        }
      }
    }
  }
}

// ---------------------------------------------------------------------------
// Kernel 3: flash attention. One wave = 16 query rows of one (batch,head) b.
// All 8 waves of a block share b, so each 32-key K/V^T block is staged ONCE
// per workgroup into double-buffered LDS via async copies: block i+1 streams
// in while WMMAs/softmax run on block i.
// Computes S^T = K*Q^T so that exp(S^T) registers are already in the exact
// B-operand packing for O^T += V^T * P^T (no LDS transpose needed).
// ---------------------------------------------------------------------------
static constexpr int KST = 72;  // K tile stride (ushorts) = 144B
static constexpr int VST = 40;  // V tile stride (ushorts) = 80B
static constexpr int KSZ = 32 * KST;
static constexpr int VSZ = 64 * VST;

__global__ __launch_bounds__(256) void attn_kernel(const __bf16* __restrict__ Q,
                                                   const __bf16* __restrict__ K,
                                                   const __bf16* __restrict__ Vt,
                                                   __bf16* __restrict__ O) {
  __shared__ __align__(16) __bf16 sK[2 * KSZ];  // 32 keys x 64 d
  __shared__ __align__(16) __bf16 sV[2 * VSZ];  // 64 d x 32 keys (V^T)

  const int tid = threadIdx.x;
  const int lane = tid & 31;
  const int w = tid >> 5;
  const int b = blockIdx.y;                   // 0..63  (= n*16 + head)
  const int r0 = blockIdx.x * 128 + w * 16;   // query row tile
  const int m = lane & 15, half = lane >> 4;

  const __bf16* Qb = Q + (size_t)b * 2048 * 64;
  const __bf16* Kb = K + (size_t)b * 2048 * 64;
  const __bf16* Vb = Vt + (size_t)b * 64 * 2048;

  // Per-thread staging coordinates (4KB per tile = 256 x 16B).
  const int kr = tid >> 3, kc = (tid & 7) * 8;  // K: 32 rows x 8 chunks
  const int vr = tid >> 2, vc = (tid & 3) * 8;  // V: 64 rows x 4 chunks

  // 2 async copies per thread per 32-key block.
  auto stage = [&](int buf, int key0) {
    async_g2l_b128(sK + buf * KSZ + kr * KST + kc,
                   Kb + (size_t)(key0 + kr) * 64 + kc);
    async_g2l_b128(sV + buf * VSZ + vr * VST + vc,
                   Vb + (size_t)vr * 2048 + key0 + vc);
  };

  // Q fragments (used as B operand: column n = query row n), d split in 2x32
  v16bf qf[2];
#pragma unroll
  for (int c = 0; c < 2; ++c)
    qf[c] = load_frag(Qb + (size_t)r0 * 64 + c * 32, 64, lane);

  v8f ot[4] = {};            // O^T tiles: d = j*16 + v + half*8, col = query m
  float mrow = -1e30f;       // running max for query row m (dup in lane m+16)
  float lrow = 0.0f;         // running sum

  stage(0, 0);
  for (int kb = 0; kb < 64; ++kb) {
    const int cur = kb & 1;
    if (kb + 1 < 64) {
      stage(cur ^ 1, (kb + 1) * 32);  // prefetch next block into other buffer
      wait_async<2>();                // ensure block `cur` has landed
    } else {
      wait_async<0>();
    }
    __syncthreads();

    const __bf16* cK = sK + cur * KSZ;
    const __bf16* cV = sV + cur * VSZ;

    // K fragments as A operand: 2 key tiles x 2 d-chunks (from LDS)
    v16bf kf[2][2];
#pragma unroll
    for (int t = 0; t < 2; ++t)
#pragma unroll
      for (int c = 0; c < 2; ++c)
        kf[t][c] = load_frag(cK + (t * 16) * KST + c * 32, KST, lane);

    // S^T tiles: row = key, col = query. Q already scaled by 1/8.
    v8f st[2];
#pragma unroll
    for (int t = 0; t < 2; ++t) {
      v8f z = {};
      z = wmma_bf16(kf[t][0], qf[0], z);
      st[t] = wmma_bf16(kf[t][1], qf[1], z);
    }

    // V^T fragments as A operand: rows = d, contiguous keys (from LDS)
    v16bf vf[4];
#pragma unroll
    for (int j = 0; j < 4; ++j)
      vf[j] = load_frag(cV + (j * 16) * VST, VST, lane);

    // online softmax: each lane holds 16 scores of its query row; partner lane
    // (xor 16) holds the other 16 keys of this 32-key block.
    float vmax = -1e30f;
#pragma unroll
    for (int t = 0; t < 2; ++t)
#pragma unroll
      for (int v = 0; v < 8; ++v) vmax = fmaxf(vmax, st[t][v]);
    vmax = fmaxf(vmax, __shfl_xor(vmax, 16, 32));
    const float mnew = fmaxf(mrow, vmax);
    const float alpha = __expf(mrow - mnew);

    // P^T in registers, repacked directly into B-operand layout:
    // element e<8 -> tile0[e], e>=8 -> tile1[e-8] (both lane halves).
    v16bf pb;
    float psum = 0.0f;
#pragma unroll
    for (int v = 0; v < 8; ++v) {
      const float p0 = __expf(st[0][v] - mnew);
      const float p1 = __expf(st[1][v] - mnew);
      psum += p0 + p1;
      pb[v] = tobf(p0);
      pb[v + 8] = tobf(p1);
    }
    psum += __shfl_xor(psum, 16, 32);
    lrow = lrow * alpha + psum;
    mrow = mnew;

#pragma unroll
    for (int j = 0; j < 4; ++j) {
      ot[j] *= alpha;
      ot[j] = wmma_bf16(vf[j], pb, ot[j]);
    }
    __syncthreads();  // all waves done with buffer `cur` before it is re-staged
  }

  // Normalize and scatter into (l, n, h*64 + d) bf16 layout for the out GEMM.
  const float inv = 1.0f / lrow;
  const int lseq = r0 + m;
  const int n = b >> 4, hh = b & 15;
  __bf16* orow = O + ((size_t)lseq * 4 + n) * 1024 + hh * 64;
#pragma unroll
  for (int j = 0; j < 4; ++j)
#pragma unroll
    for (int v = 0; v < 8; ++v)
      orow[j * 16 + v + half * 8] = tobf(ot[j][v] * inv);
}

// ---------------------------------------------------------------------------
// Launch
// ---------------------------------------------------------------------------
extern "C" void kernel_launch(void* const* d_in, const int* in_sizes, int n_in,
                              void* d_out, int out_size, void* d_ws, size_t ws_size,
                              hipStream_t stream) {
  const float* query = (const float*)d_in[0];     // (2048, 4, 1024)
  const float* qkv_proj = (const float*)d_in[1];  // (3072, 1024)
  const float* qkv_bias = (const float*)d_in[2];  // (3072,)
  const float* out_proj = (const float*)d_in[3];  // (1024, 1024)
  const float* out_bias = (const float*)d_in[4];  // (1024,)
  float* out = (float*)d_out;                     // (2048, 4, 1024) f32
  (void)in_sizes; (void)n_in; (void)out_size; (void)ws_size;

  char* ws = (char*)d_ws;
  size_t off = 0;
  auto carve = [&](size_t bytes) {
    void* p = ws + off;
    off += (bytes + 255) & ~(size_t)255;
    return p;
  };
  __bf16* qA   = (__bf16*)carve(8192ull * 1024 * 2);       // query bf16
  __bf16* wQKV = (__bf16*)carve(3072ull * 1024 * 2);       // qkv_proj bf16
  __bf16* wOUT = (__bf16*)carve(1024ull * 1024 * 2);       // out_proj bf16
  __bf16* Qb   = (__bf16*)carve(64ull * 2048 * 64 * 2);    // Q (b,l,d), pre-scaled
  __bf16* Kb   = (__bf16*)carve(64ull * 2048 * 64 * 2);    // K (b,l,d)
  __bf16* Vtb  = (__bf16*)carve(64ull * 64 * 2048 * 2);    // V^T (b,d,l)
  __bf16* Ob   = (__bf16*)carve(8192ull * 1024 * 2);       // attn out (l*4+n, i)

  cvt_f32_bf16<<<(8192 * 1024 / 4 + 255) / 256, 256, 0, stream>>>(query, qA,
                                                                  8192 * 1024 / 4);
  cvt_f32_bf16<<<(3072 * 1024 / 4 + 255) / 256, 256, 0, stream>>>(qkv_proj, wQKV,
                                                                  3072 * 1024 / 4);
  cvt_f32_bf16<<<(1024 * 1024 / 4 + 255) / 256, 256, 0, stream>>>(out_proj, wOUT,
                                                                  1024 * 1024 / 4);

  // QKV projection: M=8192, N=3072 -> grid (24, 128)
  gemm_nt<0><<<dim3(24, 128), 256, 0, stream>>>(qA, wQKV, qkv_bias, nullptr, Qb, Kb,
                                                Vtb);
  // Attention: 64 head-batches x 16 row-groups of 128 rows
  attn_kernel<<<dim3(16, 64), 256, 0, stream>>>(Qb, Kb, Vtb, Ob);
  // Output projection: M=8192, N=1024 -> grid (8, 128)
  gemm_nt<1><<<dim3(8, 128), 256, 0, stream>>>(Ob, wOUT, out_bias, out, nullptr,
                                               nullptr, nullptr);
}
